// SRU_49314814493293
// MI455X (gfx1250) — matrix operand order repaired and checked by reading, not compiled, not run on hardware
//
#include <hip/hip_runtime.h>
#include <hip/hip_bf16.h>

// ---- problem dims (fixed by the reference) ----
#define B_DIM   32
#define S_DIM   1024
#define I_DIM   1024
#define H_DIM   1024
#define M_TOTAL (B_DIM * S_DIM)   // 32768
#define K_TOTAL I_DIM             // 1024
#define N_TOTAL (3 * H_DIM)       // 3072

// ---- GEMM tiling ----
#define BM 128
#define BN 64
#define BK 32
#define NK (K_TOTAL / BK)         // 32 K-steps
#define ASTRIDE 40   // bf16 elements per A-tile row (32 data + 8 pad -> 80B)
#define WSTRIDE 40   // same for W tile

typedef __attribute__((ext_vector_type(16))) __bf16 v16bf;
typedef __attribute__((ext_vector_type(8)))  float  v8f;
typedef __attribute__((ext_vector_type(4)))  int    v4i;

// address-space-qualified v4i pointers for the async builtin
typedef __attribute__((address_space(1))) v4i as1_v4i;  // global
typedef __attribute__((address_space(3))) v4i as3_v4i;  // LDS

// ---------------- CDNA5 async global->LDS copy (ASYNCcnt path) ----------------
__device__ __forceinline__ void async_cp_b128(const unsigned short* gsrc,
                                              unsigned short* ldst) {
#if __has_builtin(__builtin_amdgcn_global_load_async_to_lds_b128)
  __builtin_amdgcn_global_load_async_to_lds_b128(
      (as1_v4i*)gsrc, (as3_v4i*)ldst, /*offset=*/0, /*cpol=*/0);
#else
  // On AMDGPU the low 32 bits of a generic LDS pointer are the LDS byte offset.
  unsigned lds_off = (unsigned)(unsigned long long)ldst;
  unsigned long long ga = (unsigned long long)gsrc;
  asm volatile("global_load_async_to_lds_b128 %0, %1, off"
               :: "v"(lds_off), "v"(ga)
               : "memory");
#endif
}

#if __has_builtin(__builtin_amdgcn_s_wait_asynccnt)
#define WAIT_ASYNC(n) __builtin_amdgcn_s_wait_asynccnt(n)
#else
#define WAIT_ASYNC(n) asm volatile("s_wait_asynccnt %0" :: "i"(n) : "memory")
#endif

__device__ __forceinline__ unsigned short f2bf(float f) {
  // round-to-nearest-even f32 -> bf16
  unsigned int u = __float_as_uint(f);
  unsigned int r = u + 0x7FFFu + ((u >> 16) & 1u);
  return (unsigned short)(r >> 16);
}

// ------------------------------------------------------------------
// Kernel 0: pack f32 -> bf16 (2 elements per 32-bit store), grid-stride
// ------------------------------------------------------------------
__global__ void sru_cvt_bf16(const float* __restrict__ src,
                             unsigned int* __restrict__ dst, int n2) {
  int i = blockIdx.x * blockDim.x + threadIdx.x;
  int stride = gridDim.x * blockDim.x;
  for (; i < n2; i += stride) {
    unsigned int lo = f2bf(src[2 * i]);
    unsigned int hi = f2bf(src[2 * i + 1]);
    dst[i] = lo | (hi << 16);
  }
}

// ------------------------------------------------------------------
// Kernel 1: wx = x @ w^T + b  via v_wmma_f32_16x16x32_bf16
//   A = xb  (M x K, row-major bf16)
//   W = wb  (N x K, row-major bf16)  => B-matrix element B[k][n] = W[n][k]
//   C = wx  (M x N, f32)
// Block: 256 threads (8 waves). Macro tile BM x BN, K chunk BK=32.
// Double-buffered LDS, staged by GLOBAL_LOAD_ASYNC_TO_LDS_B128:
// async loads complete in order, so after issuing the 3 next-tile ops,
// s_wait_asynccnt<=3 guarantees the current tile's 3 ops have landed,
// letting WMMA on tile k overlap the fetch of tile k+1.
// ------------------------------------------------------------------
__global__ __launch_bounds__(256)
void sru_gemm_bf16_wmma(const unsigned short* __restrict__ Abf,
                        const unsigned short* __restrict__ Wbf,
                        const float* __restrict__ bias,
                        float* __restrict__ Cout) {
  __shared__ unsigned short sA[2][BM * ASTRIDE]; // 2 x 10240 B
  __shared__ unsigned short sW[2][BN * WSTRIDE]; // 2 x  5120 B

  const int tid  = threadIdx.x;
  const int wave = tid >> 5;
  const int lane = tid & 31;
  const int half = lane >> 4;   // which 16-lane half of the wave
  const int l16  = lane & 15;

  const int blockN = blockIdx.x * BN;
  const int blockM = blockIdx.y * BM;
  const int waveM  = wave * 16;

  // per-thread staging coordinates (16B chunks)
  const int rowA0 = tid >> 2,           kcA0 = tid & 3;          // chunks 0..255
  const int rowA1 = (tid + 256) >> 2,   kcA1 = tid & 3;          // chunks 256..511
  const int rowW  = tid >> 2,           kcW  = tid & 3;          // chunks 0..255
  const unsigned short* gA0 = Abf + (size_t)(blockM + rowA0) * K_TOTAL + kcA0 * 8;
  const unsigned short* gA1 = Abf + (size_t)(blockM + rowA1) * K_TOTAL + kcA1 * 8;
  const unsigned short* gW  = Wbf + (size_t)(blockN + rowW ) * K_TOTAL + kcW  * 8;
  const int dA0 = rowA0 * ASTRIDE + kcA0 * 8;
  const int dA1 = rowA1 * ASTRIDE + kcA1 * 8;
  const int dW  = rowW  * WSTRIDE + kcW  * 8;

  // Accumulators (C layout: VGPR r -> row half*8 + r, col = l16), init = bias
  v8f acc[4];
#pragma unroll
  for (int j = 0; j < 4; ++j) {
    float bval = bias[blockN + j * 16 + l16];
#pragma unroll
    for (int r = 0; r < 8; ++r) acc[j][r] = bval;
  }

  // prologue: stage tile 0 into buffer 0
  async_cp_b128(gA0, &sA[0][dA0]);
  async_cp_b128(gA1, &sA[0][dA1]);
  async_cp_b128(gW,  &sW[0][dW ]);

  for (int kc = 0; kc < NK; ++kc) {
    const int cur = kc & 1;
    const bool more = (kc + 1) < NK;
    if (more) {
      const int k1 = (kc + 1) * BK;
      async_cp_b128(gA0 + k1, &sA[cur ^ 1][dA0]);
      async_cp_b128(gA1 + k1, &sA[cur ^ 1][dA1]);
      async_cp_b128(gW  + k1, &sW[cur ^ 1][dW ]);
      WAIT_ASYNC(3);  // in-order completion => tile `cur` has fully landed
    } else {
      WAIT_ASYNC(0);
    }
    __syncthreads();  // tile `cur` visible to all waves; next tile still in flight

    // ---- A fragment (ISA 16-bit A 16x32 layout):
    // lane: m = l16; VGPR0-3 = A[m][half*8 + 0..7], VGPR4-7 = A[m][16 + half*8 + 0..7]
    v16bf afrag;
    {
      v4i* ap = (v4i*)&afrag;
      const unsigned short* base = &sA[cur][(waveM + l16) * ASTRIDE];
      ap[0] = *(const v4i*)&base[half * 8];
      ap[1] = *(const v4i*)&base[16 + half * 8];
    }

    // ---- 4 B fragments + WMMA
    // B 32x16 layout: lane n = l16; VGPR0-3 = W[n][half*8 + 0..7], VGPR4-7 = W[n][16 + half*8 + 0..7]
    // -> same k order as the A fragment
#pragma unroll
    for (int j = 0; j < 4; ++j) {
      v16bf bfrag;
      v4i* bp = (v4i*)&bfrag;
      const unsigned short* base = &sW[cur][(j * 16 + l16) * WSTRIDE + half * 8];
      bp[0] = *(const v4i*)&base[0];
      bp[1] = *(const v4i*)&base[16];
      acc[j] = __builtin_amdgcn_wmma_f32_16x16x32_bf16(
          /*neg_a=*/false, afrag, /*neg_b=*/false, bfrag,
          /*c_mod=*/(short)0, acc[j], /*reuse_a=*/false, /*reuse_b=*/false);
    }
    __syncthreads();  // all waves done reading `cur` before it is restaged
  }

  // ---- epilogue: C layout store ----
#pragma unroll
  for (int j = 0; j < 4; ++j) {
    int col = blockN + j * 16 + l16;
#pragma unroll
    for (int r = 0; r < 8; ++r) {
      int row = blockM + waveM + half * 8 + r;
      Cout[(size_t)row * N_TOTAL + col] = acc[j][r];
    }
  }
}

// ------------------------------------------------------------------
// Kernel 2: sequential SRU scan over S. One thread per (b, h) channel.
// Consecutive threads -> consecutive h -> fully coalesced row reads.
// Latency-bound serial chain: prefetch the next row's lines (global_prefetch_b8).
// ------------------------------------------------------------------
__global__ __launch_bounds__(256)
void sru_scan(const float* __restrict__ wx,   // (B*S) x 3H  [f | c | r]
              const float* __restrict__ x,    // (B*S) x I   (I == H)
              const float* __restrict__ v_f,  // H
              const float* __restrict__ v_r,  // H
              float* __restrict__ hout,       // (B*S) x H
              float* __restrict__ cfin) {     // B x H
  int t = blockIdx.x * blockDim.x + threadIdx.x;
  if (t >= B_DIM * H_DIM) return;
  const int b = t / H_DIM;
  const int h = t % H_DIM;
  const float vf = v_f[h];
  const float vr = v_r[h];
  const float SCALE = 1.7320508075688772f; // sqrt(3)
  float c = 0.0f;
  const size_t rowbase = (size_t)b * S_DIM;
  for (int s = 0; s < S_DIM; ++s) {
    const size_t row = rowbase + s;
    const float* wrow = wx + row * (size_t)N_TOTAL;
    if (s + 1 < S_DIM) {  // hide next-step load latency behind this step's chain
      const float* wnext = wrow + N_TOTAL;
      __builtin_prefetch(wnext + h, 0, 0);
      __builtin_prefetch(wnext + H_DIM + h, 0, 0);
      __builtin_prefetch(wnext + 2 * H_DIM + h, 0, 0);
      __builtin_prefetch(x + (row + 1) * (size_t)H_DIM + h, 0, 0);
    }
    float wf = wrow[h];
    float wc = wrow[H_DIM + h];
    float wr = wrow[2 * H_DIM + h];
    float f  = 1.0f / (1.0f + __expf(-(wf + vf * c)));
    float cn = f * (c - wc) + wc;
    float r  = 1.0f / (1.0f + __expf(-(wr + vr * c)));  // uses prev c
    float xs = x[row * (size_t)H_DIM + h] * SCALE;
    hout[row * (size_t)H_DIM + h] = r * (cn - xs) + xs;
    c = cn;
  }
  cfin[t] = c;
}

// ------------------------------------------------------------------
extern "C" void kernel_launch(void* const* d_in, const int* in_sizes, int n_in,
                              void* d_out, int out_size, void* d_ws, size_t ws_size,
                              hipStream_t stream) {
  const float* x   = (const float*)d_in[0]; // B,S,I
  const float* w   = (const float*)d_in[1]; // 3H,I
  const float* b   = (const float*)d_in[2]; // 3H
  const float* v_f = (const float*)d_in[3]; // H
  const float* v_r = (const float*)d_in[4]; // H
  float* out = (float*)d_out;               // h (B,S,H) then c_final (B,H)

  // workspace layout
  unsigned short* xb = (unsigned short*)d_ws;                 // M*K bf16
  unsigned short* wb = xb + (size_t)M_TOTAL * K_TOTAL;        // N*K bf16
  float* wx = (float*)(wb + (size_t)N_TOTAL * K_TOTAL);       // M*N f32

  // 1) pack to bf16
  sru_cvt_bf16<<<2048, 256, 0, stream>>>(x, (unsigned int*)xb,
                                         (M_TOTAL * K_TOTAL) / 2);
  sru_cvt_bf16<<<512, 256, 0, stream>>>(w, (unsigned int*)wb,
                                        (N_TOTAL * K_TOTAL) / 2);

  // 2) wx = x @ w^T + b   (bf16 WMMA, f32 accumulate, async double-buffered)
  dim3 grid(N_TOTAL / BN, M_TOTAL / BM); // 48 x 256
  sru_gemm_bf16_wmma<<<grid, 256, 0, stream>>>(xb, wb, b, wx);

  // 3) recurrence + output gating
  sru_scan<<<(B_DIM * H_DIM) / 256, 256, 0, stream>>>(
      wx, x, v_f, v_r, out, out + (size_t)B_DIM * S_DIM * H_DIM);
}
